// GCN_71055938945675
// MI455X (gfx1250) — matrix operand order repaired and checked by reading, not compiled, not run on hardware
//
#include <hip/hip_runtime.h>

typedef __attribute__((ext_vector_type(16))) __bf16       v16bf;
typedef __attribute__((ext_vector_type(8)))  float        v8f;
typedef __attribute__((ext_vector_type(4)))  float        f32x4;
typedef __attribute__((ext_vector_type(4)))  unsigned int u32x4;
typedef __attribute__((ext_vector_type(2)))  unsigned int u32x2;

union FragBF {
    v16bf v;
    u32x4 q[2];
};

#define NROWS 8192
#define NCOLS 256
#define BM 128
#define BN 64
#define BK 64
#define LDA_S 72   // 64 + 8 pad (ushorts)
#define LDB_S 72

__device__ __forceinline__ unsigned short f2bf(float f) {
    unsigned int x = __float_as_uint(f);
    x += 0x7FFFu + ((x >> 16) & 1u);           // round-to-nearest-even
    return (unsigned short)(x >> 16);
}
__device__ __forceinline__ float bf2f(unsigned short u) {
    return __uint_as_float(((unsigned int)u) << 16);
}
__device__ __forceinline__ unsigned int hash3(unsigned int a, unsigned int b, unsigned int c) {
    unsigned int h = a * 0x9E3779B1u + b * 0x85EBCA77u + c * 0xC2B2AE3Du + 0x27220A95u;
    h ^= h >> 16; h *= 0x7FEB352Du;
    h ^= h >> 15; h *= 0x846CA68Bu;
    h ^= h >> 16;
    return h;
}

// -------- degree pass only (fallback): d[i] = rsqrt(sum_j adj[i][j] + 1) --------
__global__ __launch_bounds__(256) void deg_kernel(const float* __restrict__ adj,
                                                  float* __restrict__ dvec) {
    __shared__ float red[256];
    const int row = blockIdx.x;
    const f32x4* p = reinterpret_cast<const f32x4*>(adj + (size_t)row * NROWS);
    float s = 0.f;
    for (int c = threadIdx.x; c < NROWS / 4; c += 256) {
        f32x4 v = p[c];
        s += v[0] + v[1] + v[2] + v[3];
    }
    red[threadIdx.x] = s;
    __syncthreads();
    for (int o = 128; o > 0; o >>= 1) {
        if (threadIdx.x < o) red[threadIdx.x] += red[threadIdx.x + o];
        __syncthreads();
    }
    if (threadIdx.x == 0) dvec[row] = rsqrtf(red[0] + 1.0f);
}

// -------- fused degree + adjacency f32->bf16 conversion (single HBM pass) --------
__global__ __launch_bounds__(256) void deg_conv_kernel(const float* __restrict__ adj,
                                                       unsigned short* __restrict__ adjBf,
                                                       float* __restrict__ dvec) {
    __shared__ float red[256];
    const int row = blockIdx.x;
    const f32x4* p = reinterpret_cast<const f32x4*>(adj + (size_t)row * NROWS);
    u32x2* q = reinterpret_cast<u32x2*>(adjBf + (size_t)row * NROWS);
    float s = 0.f;
    for (int c = threadIdx.x; c < NROWS / 4; c += 256) {
        f32x4 v = p[c];
        s += v[0] + v[1] + v[2] + v[3];
        u32x2 o;
        o[0] = (unsigned int)f2bf(v[0]) | ((unsigned int)f2bf(v[1]) << 16);
        o[1] = (unsigned int)f2bf(v[2]) | ((unsigned int)f2bf(v[3]) << 16);
        q[c] = o;
    }
    red[threadIdx.x] = s;
    __syncthreads();
    for (int o = 128; o > 0; o >>= 1) {
        if (threadIdx.x < o) red[threadIdx.x] += red[threadIdx.x + o];
        __syncthreads();
    }
    if (threadIdx.x == 0) dvec[row] = rsqrtf(red[0] + 1.0f);
}

// -------- W transpose + bf16 + K pad: Wt[n][k] = bf16(W[k][n]) --------
__global__ __launch_bounds__(256) void wt_kernel(const float* __restrict__ W,
                                                 unsigned short* __restrict__ Wt,
                                                 int Kdim, int Ktot) {
    int idx = blockIdx.x * 256 + threadIdx.x;        // over 256*Ktot
    if (idx >= NCOLS * Ktot) return;
    int n = idx / Ktot;
    int k = idx - n * Ktot;
    float v = (k < Kdim) ? W[(size_t)k * NCOLS + n] : 0.f;
    Wt[(size_t)n * Ktot + k] = f2bf(v);
}

// -------- fused WMMA GEMM: C[M=8192, N=256] = A[MxK] @ B[bf16, transposed NxK]
// EPI 0: BtOut[n][i] = bf16(d[i] * acc)                       (feature GEMM)
// EPI 1: H[i][n] = dropout(relu(d[i]*(acc + B[n][i])))        (GCN layer 1/2)
// EPI 2: H[i][n] = relu(d[i]*(acc + B[n][i]))                 (GCN layer 3)
// AMODE 0: A = f32, Kdim == Ktot (no guards)
// AMODE 1: A = f32, ragged K (guards only on last tile, uniform branch)
// AMODE 2: A = bf16, Kdim == Ktot (pure 16B copies into LDS)
template <int EPI, int AMODE>
__global__ __launch_bounds__(256) void gemm_wmma(const void* __restrict__ Av, int lda,
                                                 int Kdim, int Ktot,
                                                 const unsigned short* __restrict__ Bop, // [256][Ktot]
                                                 const float* __restrict__ dvec,
                                                 unsigned short* __restrict__ BtOut,    // EPI 0
                                                 float* __restrict__ Hout,              // EPI 1/2
                                                 unsigned int seed) {
    __shared__ unsigned short As[BM * LDA_S];
    __shared__ unsigned short Bs[BN * LDB_S];

    const float* Af          = (const float*)Av;
    const unsigned short* Ab = (const unsigned short*)Av;

    const int tid  = threadIdx.x;
    const int wave = tid >> 5;
    const int lane = tid & 31;
    const int bm = blockIdx.x * BM;
    const int bn = blockIdx.y * BN;
    const int waveRow = wave * 16;
    const int l16 = lane & 15;
    const bool hiHalf = (lane >= 16);

    v8f acc[4];
#pragma unroll
    for (int t = 0; t < 4; ++t) acc[t] = 0.f;

    // staging coordinates (f32: 4-float chunks, bf16: 8-ushort chunks)
    const int aRowF = tid >> 4;            // f32: base row, +16 per i
    const int aC4   = (tid & 15) * 4;

    const int nK = Ktot / BK;
    for (int kt = 0; kt < nK; ++kt) {
        const int k0 = kt * BK;

        // ---- stage A tile (BM x BK) into bf16 LDS ----
        if (AMODE == 2) {
            // 128 rows x 64 ushorts = 1024 x 16B chunks; 4 per thread
#pragma unroll
            for (int i = 0; i < 4; ++i) {
                int c   = tid + i * 256;           // 0..1023
                int r   = c >> 3;
                int off = (c & 7) * 8;
                u32x4 v = *reinterpret_cast<const u32x4*>(Ab + (size_t)(bm + r) * lda + k0 + off);
                *reinterpret_cast<u32x4*>(&As[r * LDA_S + off]) = v;
            }
        } else if (AMODE == 0 || kt + 1 < nK) {    // uniform branch: full tile, no guards
#pragma unroll
            for (int i = 0; i < 8; ++i) {
                const int r = aRowF + i * 16;      // 0..127
                f32x4 v = *reinterpret_cast<const f32x4*>(Af + (size_t)(bm + r) * lda + (k0 + aC4));
                unsigned short* dst = &As[r * LDA_S + aC4];
                dst[0] = f2bf(v[0]); dst[1] = f2bf(v[1]);
                dst[2] = f2bf(v[2]); dst[3] = f2bf(v[3]);
            }
        } else {                                   // ragged last tile only
#pragma unroll
            for (int i = 0; i < 8; ++i) {
                const int r = aRowF + i * 16;
                const size_t gbase = (size_t)(bm + r) * lda + (k0 + aC4);
                f32x4 v;
                if (k0 + aC4 + 3 < Kdim) {
                    v = *reinterpret_cast<const f32x4*>(Af + gbase);
                } else {
#pragma unroll
                    for (int e = 0; e < 4; ++e)
                        v[e] = (k0 + aC4 + e < Kdim) ? Af[gbase + e] : 0.f;
                }
                unsigned short* dst = &As[r * LDA_S + aC4];
                dst[0] = f2bf(v[0]); dst[1] = f2bf(v[1]);
                dst[2] = f2bf(v[2]); dst[3] = f2bf(v[3]);
            }
        }
        // ---- stage B tile (BN x BK bf16, already transposed in global) ----
#pragma unroll
        for (int i = 0; i < 2; ++i) {
            int cidx = tid + i * 256;              // 0..511 16B-chunks
            int n    = cidx >> 3;
            int off  = (cidx & 7) * 8;
            u32x4 q = *reinterpret_cast<const u32x4*>(Bop + (size_t)(bn + n) * Ktot + k0 + off);
            *reinterpret_cast<u32x4*>(&Bs[n * LDB_S + off]) = q;
        }
        // prefetch next A tile (global_prefetch_b8)
        if (kt + 1 < nK) {
            if (AMODE == 2) {
                const unsigned short* pf = Ab + (size_t)(bm + (tid >> 1)) * lda + k0 + BK + (tid & 1) * 32;
                __builtin_prefetch(pf, 0, 0);
            } else {
                const float* pf = Af + (size_t)(bm + (tid >> 1)) * lda + k0 + BK + (tid & 1) * 32;
                __builtin_prefetch(pf, 0, 0);
            }
        }
        __syncthreads();

        // ---- compute: 2 sub-steps of K=32 WMMA ----
#pragma unroll
        for (int sub = 0; sub < 2; ++sub) {
            const int kb = sub * 32;
            // A fragment: lanes 0-15 -> K kb+0..7 / kb+16..23; lanes 16-31 -> +8 / +24
            const int arow = waveRow + l16;
            const int ka = kb + (hiHalf ? 8 : 0);
            FragBF afr;
            afr.q[0] = *reinterpret_cast<const u32x4*>(&As[arow * LDA_S + ka]);
            afr.q[1] = *reinterpret_cast<const u32x4*>(&As[arow * LDA_S + ka + 16]);
            const int kbb = kb + (hiHalf ? 8 : 0);
#pragma unroll
            for (int t = 0; t < 4; ++t) {
                const int nl = t * 16 + l16;
                FragBF bfr;
                bfr.q[0] = *reinterpret_cast<const u32x4*>(&Bs[nl * LDB_S + kbb]);
                bfr.q[1] = *reinterpret_cast<const u32x4*>(&Bs[nl * LDB_S + kbb + 16]);
                acc[t] = __builtin_amdgcn_wmma_f32_16x16x32_bf16(
                    false, afr.v, false, bfr.v, (short)0, acc[t], false, false);
            }
        }
        __syncthreads();
    }

    // ---- epilogue ----
    // C layout: acc[t] VGPR r -> row = bm + waveRow + r + (hiHalf?8:0), col = bn + t*16 + l16
    const int rbase = bm + waveRow + (hiHalf ? 8 : 0);
#pragma unroll
    for (int r = 0; r < 8; ++r) {
        const int i = rbase + r;
        const float di = dvec[i];
#pragma unroll
        for (int t = 0; t < 4; ++t) {
            const int n = bn + t * 16 + l16;
            const float a = acc[t][r];
            if (EPI == 0) {
                BtOut[(size_t)n * NROWS + i] = f2bf(di * a);
            } else {
                const float bi = bf2f(Bop[(size_t)n * Ktot + i]);   // identity term d[i]*M[i][n]
                float v = di * (a + bi);
                v = fmaxf(v, 0.f);
                if (EPI == 1) {
                    // dropout p=0.3: keep if hash < 0.7 * 2^32, scale by 1/0.7
                    unsigned int h = hash3(seed, (unsigned)i, (unsigned)n);
                    v = (h < 3006477107u) ? v * (1.0f / 0.7f) : 0.f;
                }
                Hout[(size_t)i * NCOLS + n] = v;
            }
        }
    }
}

// -------- logits + softmax: out[i] = softmax(H3[i] @ Wout + bout) --------
__global__ __launch_bounds__(256) void out_kernel(const float* __restrict__ H3,
                                                  const float* __restrict__ Wout, // [256][40]
                                                  const float* __restrict__ bout, // [40]
                                                  float* __restrict__ out) {      // [8192][40]
    const int wave = threadIdx.x >> 5;
    const int lane = threadIdx.x & 31;
    const int row  = blockIdx.x * 8 + wave;
    const float* h = H3 + (size_t)row * NCOLS;
    float a0 = 0.f, a1 = 0.f;
    for (int k = 0; k < NCOLS; ++k) {
        const float hv = h[k];
        a0 += hv * Wout[k * 40 + lane];
        if (lane < 8) a1 += hv * Wout[k * 40 + 32 + lane];
    }
    a0 += bout[lane];
    if (lane < 8) a1 += bout[32 + lane];

    float m = fmaxf(a0, (lane < 8) ? a1 : -3.4e38f);
    for (int o = 16; o > 0; o >>= 1) m = fmaxf(m, __shfl_xor(m, o));
    const float e0 = __expf(a0 - m);
    const float e1 = (lane < 8) ? __expf(a1 - m) : 0.f;
    float s = e0 + e1;
    for (int o = 16; o > 0; o >>= 1) s += __shfl_xor(s, o);
    const float inv = 1.0f / s;
    out[(size_t)row * 40 + lane] = e0 * inv;
    if (lane < 8) out[(size_t)row * 40 + 32 + lane] = e1 * inv;
}

extern "C" void kernel_launch(void* const* d_in, const int* in_sizes, int n_in,
                              void* d_out, int out_size, void* d_ws, size_t ws_size,
                              hipStream_t stream) {
    const float* x    = (const float*)d_in[0];   // [8192,500]
    const float* adj  = (const float*)d_in[1];   // [8192,8192]
    const float* W1   = (const float*)d_in[2];   // [500,256]
    const float* W2   = (const float*)d_in[3];   // [256,256]
    const float* W3   = (const float*)d_in[4];   // [256,256]
    const float* Wout = (const float*)d_in[5];   // [256,40]
    const float* bout = (const float*)d_in[6];   // [40]
    float* out = (float*)d_out;

    char* ws = (char*)d_ws;
    float*          dvec = (float*)(ws);                       //  32 KB
    unsigned short* Bt   = (unsigned short*)(ws + 32768);      //   4 MB  [256][8192] bf16
    unsigned short* Wt   = (unsigned short*)(ws + 4227072);    // 256 KB  [256][<=512] bf16
    float*          H    = (float*)(ws + 4489216);             //   8 MB  [8192][256] f32
    unsigned short* adjB = (unsigned short*)(ws + 12877824);   // 128 MB  [8192][8192] bf16
    const size_t need_big = 12877824 + (size_t)NROWS * NROWS * 2;  // ~140 MB

    const dim3 gG(NROWS / BM, NCOLS / BN);   // (64, 4)
    const bool big = (ws_size >= need_big);

    if (big) {
        // one f32 pass: degree vector + bf16 adjacency copy (L2-resident afterwards)
        deg_conv_kernel<<<NROWS, 256, 0, stream>>>(adj, adjB, dvec);
    } else {
        deg_kernel<<<NROWS, 256, 0, stream>>>(adj, dvec);
    }

    // ---- layer 1 ----
    wt_kernel<<<512, 256, 0, stream>>>(W1, Wt, 500, 512);
    gemm_wmma<0, 1><<<gG, 256, 0, stream>>>(x, 500, 500, 512, Wt, dvec, Bt, nullptr, 0u);
    if (big) gemm_wmma<1, 2><<<gG, 256, 0, stream>>>(adjB, NROWS, NROWS, NROWS, Bt, dvec, nullptr, H, 1u);
    else     gemm_wmma<1, 0><<<gG, 256, 0, stream>>>(adj,  NROWS, NROWS, NROWS, Bt, dvec, nullptr, H, 1u);

    // ---- layer 2 ----
    wt_kernel<<<256, 256, 0, stream>>>(W2, Wt, 256, 256);
    gemm_wmma<0, 0><<<gG, 256, 0, stream>>>(H, 256, 256, 256, Wt, dvec, Bt, nullptr, 0u);
    if (big) gemm_wmma<1, 2><<<gG, 256, 0, stream>>>(adjB, NROWS, NROWS, NROWS, Bt, dvec, nullptr, H, 2u);
    else     gemm_wmma<1, 0><<<gG, 256, 0, stream>>>(adj,  NROWS, NROWS, NROWS, Bt, dvec, nullptr, H, 2u);

    // ---- layer 3 (no dropout) ----
    wt_kernel<<<256, 256, 0, stream>>>(W3, Wt, 256, 256);
    gemm_wmma<0, 0><<<gG, 256, 0, stream>>>(H, 256, 256, 256, Wt, dvec, Bt, nullptr, 0u);
    if (big) gemm_wmma<2, 2><<<gG, 256, 0, stream>>>(adjB, NROWS, NROWS, NROWS, Bt, dvec, nullptr, H, 0u);
    else     gemm_wmma<2, 0><<<gG, 256, 0, stream>>>(adj,  NROWS, NROWS, NROWS, Bt, dvec, nullptr, H, 0u);

    // ---- output head + softmax ----
    out_kernel<<<NROWS / 8, 256, 0, stream>>>(H, Wout, bout, out);
}
